// BERT_BiLSTM_CRF_62972810494586
// MI455X (gfx1250) — matrix operand order repaired and checked by reading, not compiled, not run on hardware
//
#include <hip/hip_runtime.h>
#include <hip/hip_bf16.h>

typedef __attribute__((ext_vector_type(16))) _Float16 v16h;
typedef __attribute__((ext_vector_type(8)))  _Float16 v8h;
typedef __attribute__((ext_vector_type(4)))  _Float16 v4h;
typedef __attribute__((ext_vector_type(8)))  float    v8f;
typedef __attribute__((ext_vector_type(4)))  float    v4f;

#define H     768
#define HALF  256
#define TT    20
#define S2    16384
#define NEGV  (-10000.0f)
#define TSTART 18
#define TSTOP  19

// ---- workspace layout (bytes) ----
#define WPACK_ELEMS (2*3*H*HALF)                 // f16 weights, N-major
#define BIAS_OFF    ((size_t)WPACK_ELEMS*2)      // 2,359,296
#define ENC_OFF     (BIAS_OFF + (size_t)2*3*HALF*4)
#define FEATS_OFF   (ENC_OFF + (size_t)S2*512*4)
#define BP_OFF      (FEATS_OFF + (size_t)S2*TT*4)

__device__ __forceinline__ float sigmoid_(float x) {
    return 1.0f / (1.0f + __expf(-x));
}
__device__ __forceinline__ float tanh_(float x) {
    float e = __expf(2.0f * x);
    return 1.0f - 2.0f / (e + 1.0f);
}

// ---------------------------------------------------------------------------
// Kernel 1: pack live gate blocks (i, g, o) of w_ih to f16, N-major layout
// wpack[((dir*3+g)*HALF + n)*H + k] = w_ih[dir][goff[g]+n][k]
// bias[(dir*3+g)*HALF + n] = b_ih[goff+n] + b_hh[goff+n]
// ---------------------------------------------------------------------------
__global__ void pack_weights(const float* __restrict__ wf, const float* __restrict__ wb,
                             const float* __restrict__ bihf, const float* __restrict__ bhhf,
                             const float* __restrict__ bihb, const float* __restrict__ bhhb,
                             _Float16* __restrict__ wpack, float* __restrict__ bias) {
    int idx = blockIdx.x * 256 + threadIdx.x;
    const int goff[3] = {0, 2 * HALF, 3 * HALF};   // i, g, o rows in w_ih
    if (idx < WPACK_ELEMS) {
        int dir = idx / (3 * H * HALF);
        int r   = idx % (3 * H * HALF);
        int g   = r / (H * HALF);
        int r2  = r % (H * HALF);
        int k   = r2 % H;
        int n   = r2 / H;
        const float* w = dir ? wb : wf;
        wpack[idx] = (_Float16)w[(size_t)(goff[g] + n) * H + k];
    }
    if (idx < 2 * 3 * HALF) {
        int dir = idx / (3 * HALF);
        int r   = idx % (3 * HALF);
        int g   = r / HALF;
        int n   = r % HALF;
        const float* bi = dir ? bihb : bihf;
        const float* bh = dir ? bhhb : bhhf;
        bias[idx] = bi[goff[g] + n] + bh[goff[g] + n];
    }
}

// ---------------------------------------------------------------------------
// Kernel 2: gates GEMM + gate fusion.
// Block: 256 threads = 8 waves; block tile M=64 x N=128 (one dir / N-half).
// Wave: 32(M) x 32(N) x 3 gates -> 12 accumulators (96 VGPRs), bfrag[3][2]
// (48 VGPRs): total ~190 VGPRs -> ~5 waves/SIMD so load latency is hidden
// by co-resident waves' WMMA streams.
// ---------------------------------------------------------------------------
__global__ __launch_bounds__(256)
void lstm_gates_gemm(const float* __restrict__ x,
                     const _Float16* __restrict__ wpack,
                     const float* __restrict__ bias,
                     float* __restrict__ enc) {
    __shared__ float ltok[64 * 68];               // 64 rows x 64 K, stride 68 (bank spread)

    const int tid  = threadIdx.x;
    const int lane = tid & 31;
    const int wave = tid >> 5;
    const int wm   = wave >> 2;                   // 0..1 : 32-row group
    const int wn   = wave & 3;                    // 0..3 : 32-col group
    const int mblk = blockIdx.x;                  // 0..255
    const int dir  = blockIdx.y >> 1;             // 0..1
    const int nhalf= blockIdx.y & 1;              // 0..1 : 128-col half

    const int lm  = lane & 15;
    const int lhi = lane >> 4;                    // 0/1

    const int ncol0 = nhalf * 128 + wn * 32;      // column offset within the 256 gate cols

    // biases per lane (depend only on N column)
    float bval[3][2];
#pragma unroll
    for (int g = 0; g < 3; ++g)
#pragma unroll
        for (int nt = 0; nt < 2; ++nt)
            bval[g][nt] = bias[(dir * 3 + g) * HALF + ncol0 + nt * 16 + lm];

    v8f acc[3][2][2];
#pragma unroll
    for (int g = 0; g < 3; ++g)
#pragma unroll
        for (int mt = 0; mt < 2; ++mt)
#pragma unroll
            for (int nt = 0; nt < 2; ++nt)
                acc[g][mt][nt] = 0;

    // per-lane B base: lm = N column, lhi selects K half (16 contiguous K); (g,nt,kk) are immediates
    const _Float16* bbase0 = wpack + ((size_t)(dir * 3 * HALF + ncol0 + lm)) * H + lhi * 16;

    // tok = x[0, 1:-1, :] -> row offset +1
    const float* tokbase = x + (size_t)(mblk * 64 + 1) * H;
    const int lrow  = tid >> 4;                   // 0..15
    const int lcol4 = (tid & 15) * 4;             // 0,4,...,60

    for (int k0 = 0; k0 < H; k0 += 64) {
        __syncthreads();
        // cooperative load: 64 rows x 64 K fp32 -> LDS
#pragma unroll
        for (int rr = 0; rr < 4; ++rr) {
            int row = lrow + rr * 16;
            v4f v = *(const v4f*)(tokbase + (size_t)row * H + k0 + lcol4);
            *(v4f*)(&ltok[row * 68 + lcol4]) = v;
        }
        if (k0 + 64 < H) {
            __builtin_prefetch(tokbase + (size_t)lrow * H + k0 + 64, 0, 0); // global_prefetch_b8
        }
        __syncthreads();

        const _Float16* bbase = bbase0 + (size_t)k0;

#pragma unroll
        for (int kk = 0; kk < 64; kk += 32) {
            // batch-load all B fragments for this K-step (immediate offsets)
            v16h bfrag[3][2];
#pragma unroll
            for (int g = 0; g < 3; ++g)
#pragma unroll
                for (int nt = 0; nt < 2; ++nt)
                    bfrag[g][nt] = *(const v16h*)(bbase + (g * HALF + nt * 16) * H + kk);

            // A fragments from LDS (16-bit A 16x32 layout)
            v16h afrag[2];
#pragma unroll
            for (int mt = 0; mt < 2; ++mt) {
                int row = wm * 32 + mt * 16 + lm;
                const float* p = &ltok[row * 68 + kk + lhi * 16];
                v4f f0 = *(const v4f*)(p);
                v4f f1 = *(const v4f*)(p + 4);
                v4f f2 = *(const v4f*)(p + 8);
                v4f f3 = *(const v4f*)(p + 12);
                v4h h0 = __builtin_convertvector(f0, v4h);
                v4h h1 = __builtin_convertvector(f1, v4h);
                v4h h2 = __builtin_convertvector(f2, v4h);
                v4h h3 = __builtin_convertvector(f3, v4h);
                v8h lo = __builtin_shufflevector(h0, h1, 0, 1, 2, 3, 4, 5, 6, 7);
                v8h hi = __builtin_shufflevector(h2, h3, 0, 1, 2, 3, 4, 5, 6, 7);
                afrag[mt] = __builtin_shufflevector(lo, hi, 0, 1, 2, 3, 4, 5, 6, 7,
                                                    8, 9, 10, 11, 12, 13, 14, 15);
            }

            // 12 back-to-back WMMAs
#pragma unroll
            for (int g = 0; g < 3; ++g)
#pragma unroll
                for (int nt = 0; nt < 2; ++nt)
#pragma unroll
                    for (int mt = 0; mt < 2; ++mt)
                        acc[g][mt][nt] = __builtin_amdgcn_wmma_f32_16x16x32_f16(
                            false, afrag[mt], false, bfrag[g][nt], (short)0,
                            acc[g][mt][nt], false, false);
        }
    }

    // epilogue: fused LSTM cell (f gate dead), write enc[:, dir*256 : dir*256+256]
#pragma unroll
    for (int mt = 0; mt < 2; ++mt) {
#pragma unroll
        for (int nt = 0; nt < 2; ++nt) {
            v8f vi = acc[0][mt][nt];
            v8f vg = acc[1][mt][nt];
            v8f vo = acc[2][mt][nt];
            int ncol = dir * HALF + ncol0 + nt * 16 + lm;
            int mrow = mblk * 64 + wm * 32 + mt * 16 + lhi * 8;
#pragma unroll
            for (int r = 0; r < 8; ++r) {
                float iv = sigmoid_(vi[r] + bval[0][nt]);
                float gv = tanh_(vg[r] + bval[1][nt]);
                float ov = sigmoid_(vo[r] + bval[2][nt]);
                float c  = iv * gv;
                float h  = ov * tanh_(c);
                enc[(size_t)(mrow + r) * 512 + ncol] = h;
            }
        }
    }
}

// ---------------------------------------------------------------------------
// Kernel 3: feats = enc @ w_tag.T + b_tag   (512 -> 20, tiny)
// ---------------------------------------------------------------------------
__global__ __launch_bounds__(256)
void tag_proj(const float* __restrict__ enc, const float* __restrict__ w_tag,
              const float* __restrict__ b_tag, float* __restrict__ feats) {
    __shared__ float wt[TT * 512];
    __shared__ float bt[TT];
    int tid = threadIdx.x;
    for (int i = tid; i < TT * 512; i += 256) wt[i] = w_tag[i];
    if (tid < TT) bt[tid] = b_tag[tid];
    __syncthreads();

    int row0 = blockIdx.x * 64;
    for (int idx = tid; idx < 64 * TT; idx += 256) {
        int r = idx / TT, t = idx % TT;
        const float* e = enc + (size_t)(row0 + r) * 512;
        const float* w = &wt[t * 512];
        v4f accv = 0;
        for (int k = 0; k < 512; k += 4) {
            v4f ev = *(const v4f*)(e + k);
            v4f wv = *(const v4f*)(w + k);
            accv += ev * wv;
        }
        feats[(size_t)(row0 + r) * TT + t] = bt[t] + accv[0] + accv[1] + accv[2] + accv[3];
    }
}

// ---------------------------------------------------------------------------
// Kernel 4: Viterbi forward scan + backtrack (single wave, inherently serial)
// ---------------------------------------------------------------------------
__global__ void viterbi(const float* __restrict__ feats,
                        const float* __restrict__ trans,
                        int* __restrict__ bp, float* __restrict__ out) {
    __shared__ float fv[32];
    __shared__ float nf[32];
    int j = threadIdx.x;

    float tr[TT];
    if (j < TT) {
#pragma unroll
        for (int i = 0; i < TT; ++i) tr[i] = trans[j * TT + i];
    }
    fv[j] = (j == TSTART) ? 0.0f : NEGV;
    __syncthreads();

    for (int t = 0; t < S2; ++t) {
        if (j < TT) {
            float best = -3.0e38f;
            int arg = 0;
#pragma unroll
            for (int i = 0; i < TT; ++i) {
                float s = fv[i] + tr[i];
                if (s > best) { best = s; arg = i; }   // first max wins (matches argmax)
            }
            nf[j] = best + feats[(size_t)t * TT + j];
            bp[(size_t)t * TT + j] = arg;
        }
        __syncthreads();
        if (j < TT) fv[j] = nf[j];
        __syncthreads();
    }

    nf[j] = (j < TT) ? fv[j] + trans[TSTOP * TT + j] : -3.0e38f;
    __syncthreads();

    if (j == 0) {
        int b = 0;
        float bs = nf[0];
        for (int i = 1; i < TT; ++i)
            if (nf[i] > bs) { bs = nf[i]; b = i; }
        out[0] = bs;                                   // path_score
        for (int t = S2 - 1; t >= 0; --t) {            // reverse scan backtrack
            out[1 + t] = (float)b;
            b = bp[(size_t)t * TT + b];
        }
    }
}

// ---------------------------------------------------------------------------
extern "C" void kernel_launch(void* const* d_in, const int* in_sizes, int n_in,
                              void* d_out, int out_size, void* d_ws, size_t ws_size,
                              hipStream_t stream) {
    (void)in_sizes; (void)n_in; (void)out_size; (void)ws_size;

    const float* x      = (const float*)d_in[0];
    const float* w_ih_f = (const float*)d_in[1];
    // d_in[2] = w_hh_f : unused by the reference math
    const float* b_ih_f = (const float*)d_in[3];
    const float* b_hh_f = (const float*)d_in[4];
    const float* w_ih_b = (const float*)d_in[5];
    // d_in[6] = w_hh_b : unused
    const float* b_ih_b = (const float*)d_in[7];
    const float* b_hh_b = (const float*)d_in[8];
    const float* w_tag  = (const float*)d_in[9];
    const float* b_tag  = (const float*)d_in[10];
    const float* trans  = (const float*)d_in[11];

    char* ws = (char*)d_ws;
    _Float16* wpack = (_Float16*)(ws);
    float*    bias  = (float*)(ws + BIAS_OFF);
    float*    enc   = (float*)(ws + ENC_OFF);
    float*    feats = (float*)(ws + FEATS_OFF);
    int*      bp    = (int*)(ws + BP_OFF);

    pack_weights<<<(WPACK_ELEMS + 255) / 256, 256, 0, stream>>>(
        w_ih_f, w_ih_b, b_ih_f, b_hh_f, b_ih_b, b_hh_b, wpack, bias);

    dim3 grid(S2 / 64, 4);   // x: M-blocks, y: dir * 2 + N-half
    lstm_gates_gemm<<<grid, 256, 0, stream>>>(x, wpack, bias, enc);

    tag_proj<<<S2 / 64, 256, 0, stream>>>(enc, w_tag, b_tag, feats);

    viterbi<<<1, 32, 0, stream>>>(feats, trans, bp, (float*)d_out);
}
